// fast_weights_model_24721831756480
// MI455X (gfx1250) — compile-verified, matrix operand order checked by reading, not run on hardware
//
#include <hip/hip_runtime.h>
#include <hip/hip_bf16.h>
#include <math.h>

#define BN 32768
#define TN 11
#define EN 128
#define HN 20

typedef __bf16 v16bf __attribute__((ext_vector_type(16)));
typedef float  v8f   __attribute__((ext_vector_type(8)));

union FragBF { unsigned short u[16]; v16bf v; };

// native f32 -> bf16 (RNE) conversion
static __device__ inline unsigned short f2bf(float f) {
    return __builtin_bit_cast(unsigned short, (__bf16)f);
}
static __device__ inline v8f vzero8() {
    v8f z = {0.f,0.f,0.f,0.f,0.f,0.f,0.f,0.f};
    return z;
}
// CDNA5 16-bit A/B fragment: lane half hf, element e -> K index within 32-wide chunk
static __device__ inline int kmap(int e, int hf) {
    return (e < 8) ? (8*hf + e) : (16 + 8*hf + (e - 8));
}
// load one packed fragment (32 bytes per lane) with 2x ds_load_b128
#define FRAG_LOAD(dst, base, fidx, lane)                                          \
    { const uint4* _p = (const uint4*)&(base)[(((fidx)*32 + (lane))) << 4];       \
      *(uint4*)&(dst).u[0] = _p[0]; *(uint4*)&(dst).u[8] = _p[1]; }

// ---------------------------------------------------------------------------
// Kernel 1: zc[b,t,:] = relu(relu(bx@w1+b1)@w2+b2)@c   via bf16 WMMA chain
// Weights live in LDS pre-packed in per-lane fragment order.
// ---------------------------------------------------------------------------
__global__ void __launch_bounds__(128)
zc_kernel(const float* __restrict__ bx,
          const float* __restrict__ w1, const float* __restrict__ b1,
          const float* __restrict__ w2, const float* __restrict__ b2,
          const float* __restrict__ cm,
          float* __restrict__ zc)
{
    __shared__ __align__(16) unsigned short W1p[16*512];  // 4 kc x 4 nt frags
    __shared__ __align__(16) unsigned short W2p[14*512];  // 2 kc x 7 nt frags
    __shared__ __align__(16) unsigned short Cp [ 8*512];  // 4 kc x 2 nt frags
    __shared__ float B1s[64];
    __shared__ float B2s[112];
    __shared__ __align__(16) unsigned short Scr[4][4*512]; // per-wave packed A scratch

    const int tid = threadIdx.x;
    for (int i = tid; i < 16*512; i += 128) {
        int e = i & 15, l = (i >> 4) & 31, f = i >> 9;
        int kc = f >> 2, nt = f & 3;
        int n = nt*16 + (l & 15);
        int k = kc*32 + kmap(e, l >> 4);
        W1p[i] = (n < 50) ? f2bf(w1[k*50 + n]) : (unsigned short)0;
    }
    for (int i = tid; i < 14*512; i += 128) {
        int e = i & 15, l = (i >> 4) & 31, f = i >> 9;
        int kc = f / 7, nt = f % 7;
        int n = nt*16 + (l & 15);
        int k = kc*32 + kmap(e, l >> 4);
        W2p[i] = (k < 50 && n < 100) ? f2bf(w2[k*100 + n]) : (unsigned short)0;
    }
    for (int i = tid; i < 8*512; i += 128) {
        int e = i & 15, l = (i >> 4) & 31, f = i >> 9;
        int kc = f >> 1, nt = f & 1;
        int n = nt*16 + (l & 15);
        int k = kc*32 + kmap(e, l >> 4);
        Cp[i] = (k < 100 && n < HN) ? f2bf(cm[k*HN + n]) : (unsigned short)0;
    }
    for (int i = tid; i < 64;  i += 128) B1s[i] = (i < 50)  ? b1[i] : 0.f;
    for (int i = tid; i < 112; i += 128) B2s[i] = (i < 100) ? b2[i] : 0.f;
    __syncthreads();

    const int lane = tid & 31;
    const int wave = tid >> 5;
    const int m   = lane & 15;
    const int hf  = lane >> 4;
    const int n16 = lane & 15;
    unsigned short* scr = &Scr[wave][0];

    const int nTiles = (BN * TN) / 16;
    const int stride = gridDim.x * 4;
    for (int tile = blockIdx.x*4 + wave; tile < nTiles; tile += stride) {
        // prefetch next tile's row (4 x 128B lines)
        int nxt = tile + stride;
        if (nxt < nTiles) {
            const float* pf = bx + (size_t)nxt*16*EN + (size_t)m*EN;
            __builtin_prefetch(pf,       0, 0);
            __builtin_prefetch(pf +  32, 0, 0);
            __builtin_prefetch(pf +  64, 0, 0);
            __builtin_prefetch(pf +  96, 0, 0);
        }
        const float4* xr4 = (const float4*)(bx + (size_t)tile*16*EN + (size_t)m*EN);

        // ---- layer 1: [16x128] @ [128x64] ----
        v8f acc1[4];
        #pragma unroll
        for (int nt = 0; nt < 4; ++nt) acc1[nt] = vzero8();
        #pragma unroll
        for (int kc = 0; kc < 4; ++kc) {
            FragBF a;
            float4 f0 = xr4[kc*8 + 2*hf];
            float4 f1 = xr4[kc*8 + 2*hf + 1];
            float4 f2 = xr4[kc*8 + 4 + 2*hf];
            float4 f3 = xr4[kc*8 + 4 + 2*hf + 1];
            a.u[ 0]=f2bf(f0.x); a.u[ 1]=f2bf(f0.y); a.u[ 2]=f2bf(f0.z); a.u[ 3]=f2bf(f0.w);
            a.u[ 4]=f2bf(f1.x); a.u[ 5]=f2bf(f1.y); a.u[ 6]=f2bf(f1.z); a.u[ 7]=f2bf(f1.w);
            a.u[ 8]=f2bf(f2.x); a.u[ 9]=f2bf(f2.y); a.u[10]=f2bf(f2.z); a.u[11]=f2bf(f2.w);
            a.u[12]=f2bf(f3.x); a.u[13]=f2bf(f3.y); a.u[14]=f2bf(f3.z); a.u[15]=f2bf(f3.w);
            #pragma unroll
            for (int nt = 0; nt < 4; ++nt) {
                FragBF bm;
                FRAG_LOAD(bm, W1p, kc*4 + nt, lane);
                acc1[nt] = __builtin_amdgcn_wmma_f32_16x16x32_bf16(
                    false, a.v, false, bm.v, (short)0, acc1[nt], false, false);
            }
        }
        // relu + bias, scatter into packed-A scratch (K = 0..63 for layer 2)
        #pragma unroll
        for (int nt = 0; nt < 4; ++nt) {
            int nn = nt*16 + n16;
            int kc2 = nn >> 5, kk = nn & 31;
            int hf2 = (kk >> 3) & 1;
            int e2  = (kk & 7) | ((kk >> 1) & 8);
            unsigned short* dst = &scr[((kc2*32 + hf2*16 + 8*hf) << 4) + e2];
            #pragma unroll
            for (int r = 0; r < 8; ++r)
                dst[r << 4] = f2bf(fmaxf(acc1[nt][r] + B1s[nn], 0.f));
        }
        __builtin_amdgcn_wave_barrier();

        // ---- layer 2: [16x64] @ [64x112] ----
        FragBF a2[2];
        #pragma unroll
        for (int kc = 0; kc < 2; ++kc) FRAG_LOAD(a2[kc], scr, kc, lane);
        v8f acc2[7];
        #pragma unroll
        for (int nt = 0; nt < 7; ++nt) acc2[nt] = vzero8();
        #pragma unroll
        for (int kc = 0; kc < 2; ++kc)
            #pragma unroll
            for (int nt = 0; nt < 7; ++nt) {
                FragBF bm;
                FRAG_LOAD(bm, W2p, kc*7 + nt, lane);
                acc2[nt] = __builtin_amdgcn_wmma_f32_16x16x32_bf16(
                    false, a2[kc].v, false, bm.v, (short)0, acc2[nt], false, false);
            }
        __builtin_amdgcn_wave_barrier();
        // relu + bias, scatter K = 0..111; zero-fill K = 112..127
        #pragma unroll
        for (int nt = 0; nt < 7; ++nt) {
            int nn = nt*16 + n16;
            int kc2 = nn >> 5, kk = nn & 31;
            int hf2 = (kk >> 3) & 1;
            int e2  = (kk & 7) | ((kk >> 1) & 8);
            unsigned short* dst = &scr[((kc2*32 + hf2*16 + 8*hf) << 4) + e2];
            #pragma unroll
            for (int r = 0; r < 8; ++r)
                dst[r << 4] = f2bf(fmaxf(acc2[nt][r] + B2s[nn], 0.f));
        }
        {
            uint4 z4 = make_uint4(0u, 0u, 0u, 0u);
            *(uint4*)&scr[((3*32 + lane) << 4) + 8] = z4;
        }
        __builtin_amdgcn_wave_barrier();

        // ---- layer 3: [16x128] @ [128x32] -> zc ----
        v8f acc3[2];
        acc3[0] = vzero8(); acc3[1] = vzero8();
        #pragma unroll
        for (int kc = 0; kc < 4; ++kc) {
            FragBF a3;
            FRAG_LOAD(a3, scr, kc, lane);
            #pragma unroll
            for (int nt = 0; nt < 2; ++nt) {
                FragBF bm;
                FRAG_LOAD(bm, Cp, kc*2 + nt, lane);
                acc3[nt] = __builtin_amdgcn_wmma_f32_16x16x32_bf16(
                    false, a3.v, false, bm.v, (short)0, acc3[nt], false, false);
            }
        }
        #pragma unroll
        for (int nt = 0; nt < 2; ++nt) {
            int nn = nt*16 + n16;
            if (nn < HN) {
                #pragma unroll
                for (int r = 0; r < 8; ++r) {
                    int row = tile*16 + r + 8*hf;
                    zc[(size_t)row*HN + nn] = acc3[nt][r];
                }
            }
        }
    }
}

// ---------------------------------------------------------------------------
// One recurrence step (pre-norm): consumes previous s_pre + stats, produces
// r_t history, new s_pre, and deterministic per-block (sum, sumsq) partials.
// ---------------------------------------------------------------------------
__global__ void __launch_bounds__(256)
step_kernel(const float* __restrict__ zc, const float* __restrict__ wmat,
            const float* __restrict__ lp, const float* __restrict__ ep,
            const float* __restrict__ gp, const float* __restrict__ bp,
            const float* __restrict__ stats,
            float* __restrict__ rhist, float* __restrict__ spre,
            float* __restrict__ partials, int t)
{
    __shared__ float Ws[HN*HN];
    __shared__ float Cn[4*HN];           // g, b, mu, rstd
    __shared__ float sm[256*HN];
    const int tid = threadIdx.x;
    for (int i = tid; i < HN*HN; i += 256) Ws[i] = wmat[i];
    if (tid < HN) {
        Cn[tid]        = gp[tid];
        Cn[HN + tid]   = bp[tid];
        Cn[2*HN + tid] = stats[tid];
        Cn[3*HN + tid] = stats[HN + tid];
    }
    __syncthreads();

    const float lv = lp[0], ev = ep[0];
    const int row = blockIdx.x*256 + tid;

    float hprev[HN], r[HN], acc[HN], zr[HN];
    if (t == 0) {
        #pragma unroll
        for (int j = 0; j < HN; ++j) hprev[j] = 0.f;
    } else {
        #pragma unroll
        for (int j = 0; j < HN; ++j) {
            float sp = spre[(size_t)row*HN + j];
            hprev[j] = fmaxf(Cn[j]*(sp - Cn[2*HN + j])*Cn[3*HN + j] + Cn[HN + j], 0.f);
        }
    }
    #pragma unroll
    for (int j = 0; j < HN; ++j) zr[j] = zc[((size_t)row*TN + t)*HN + j];

    #pragma unroll
    for (int j = 0; j < HN; ++j) {
        float s = zr[j];
        #pragma unroll
        for (int i = 0; i < HN; ++i) s += hprev[i]*Ws[i*HN + j];
        r[j] = fmaxf(s, 0.f);
    }
    #pragma unroll
    for (int j = 0; j < HN; ++j) rhist[((size_t)t*BN + row)*HN + j] = r[j];

    #pragma unroll
    for (int j = 0; j < HN; ++j) {
        float s = zr[j];
        #pragma unroll
        for (int i = 0; i < HN; ++i) s += r[i]*Ws[i*HN + j];
        acc[j] = s;
    }
    // fast-weight term: e * sum_{s<=t} l^{t-s} (r . r_s) r_s
    float dot = 0.f;
    #pragma unroll
    for (int i = 0; i < HN; ++i) dot += r[i]*r[i];
    float coef = ev*dot;
    #pragma unroll
    for (int j = 0; j < HN; ++j) acc[j] += coef*r[j];
    float dec = ev*lv;
    for (int s = t - 1; s >= 0; --s) {
        const float* rsp = rhist + ((size_t)s*BN + row)*HN;
        float rsv[HN];
        #pragma unroll
        for (int i = 0; i < HN; ++i) rsv[i] = rsp[i];
        float d = 0.f;
        #pragma unroll
        for (int i = 0; i < HN; ++i) d += r[i]*rsv[i];
        float cf = dec*d;
        #pragma unroll
        for (int j = 0; j < HN; ++j) acc[j] += cf*rsv[j];
        dec *= lv;
    }
    #pragma unroll
    for (int j = 0; j < HN; ++j) spre[(size_t)row*HN + j] = acc[j];

    // deterministic block partials (fixed-order column sums)
    #pragma unroll
    for (int j = 0; j < HN; ++j) sm[tid*HN + j] = acc[j];
    __syncthreads();
    if (tid < 2*HN) {
        int j = tid % HN;
        bool sq = tid >= HN;
        float s = 0.f;
        for (int k = 0; k < 256; ++k) {
            float v = sm[k*HN + j];
            s += sq ? v*v : v;
        }
        partials[blockIdx.x*(2*HN) + tid] = s;
    }
}

__global__ void stats_kernel(const float* __restrict__ partials,
                             float* __restrict__ stats, int nblk)
{
    int j = threadIdx.x;
    if (j < HN) {
        float s = 0.f, s2 = 0.f;
        for (int b = 0; b < nblk; ++b) {
            s  += partials[b*(2*HN) + j];
            s2 += partials[b*(2*HN) + HN + j];
        }
        float mu  = s / (float)BN;
        float var = fmaxf(s2 / (float)BN - mu*mu, 0.f);
        stats[j]      = mu;
        stats[HN + j] = 1.0f / sqrtf(var);
    }
}

// ---------------------------------------------------------------------------
// Head: h = norm(s_pre); h2 = relu(h@w3+b3); logits = h2@w4+b4  (WMMA)
// ---------------------------------------------------------------------------
__global__ void __launch_bounds__(128)
head_kernel(const float* __restrict__ spre, const float* __restrict__ stats,
            const float* __restrict__ gp, const float* __restrict__ bp,
            const float* __restrict__ w3, const float* __restrict__ b3,
            const float* __restrict__ w4, const float* __restrict__ b4,
            float* __restrict__ logits)
{
    __shared__ __align__(16) unsigned short W3p[ 7*512];   // 1 kc x 7 nt frags
    __shared__ __align__(16) unsigned short W4p[32*512];   // 4 kc x 8 nt frags
    __shared__ float B3s[112];
    __shared__ float B4s[128];
    __shared__ float Cn[4*HN];                // g, b, mu, rstd
    __shared__ __align__(16) unsigned short Scr[4][4*512];

    const int tid = threadIdx.x;
    for (int i = tid; i < 7*512; i += 128) {
        int e = i & 15, l = (i >> 4) & 31, nt = i >> 9;
        int n = nt*16 + (l & 15);
        int k = kmap(e, l >> 4);
        W3p[i] = (k < HN && n < 100) ? f2bf(w3[k*100 + n]) : (unsigned short)0;
    }
    for (int i = tid; i < 32*512; i += 128) {
        int e = i & 15, l = (i >> 4) & 31, f = i >> 9;
        int kc = f >> 3, nt = f & 7;
        int n = nt*16 + (l & 15);
        int k = kc*32 + kmap(e, l >> 4);
        W4p[i] = (k < 100) ? f2bf(w4[k*128 + n]) : (unsigned short)0;
    }
    for (int i = tid; i < 112; i += 128) B3s[i] = (i < 100) ? b3[i] : 0.f;
    for (int i = tid; i < 128; i += 128) B4s[i] = b4[i];
    if (tid < HN) {
        Cn[tid]        = gp[tid];
        Cn[HN + tid]   = bp[tid];
        Cn[2*HN + tid] = stats[tid];
        Cn[3*HN + tid] = stats[HN + tid];
    }
    __syncthreads();

    const int lane = tid & 31;
    const int wave = tid >> 5;
    const int m   = lane & 15;
    const int hf  = lane >> 4;
    const int n16 = lane & 15;
    unsigned short* scr = &Scr[wave][0];

    const int nTiles = BN / 16;
    for (int tile = blockIdx.x*4 + wave; tile < nTiles; tile += gridDim.x*4) {
        const int row = tile*16 + m;

        // A fragment: normalized h, K padded 20 -> 32 (single k-chunk)
        FragBF a;
        #pragma unroll
        for (int e = 0; e < 16; ++e) {
            int k = kmap(e, hf);
            float v = 0.f;
            if (k < HN) {
                float sp = spre[(size_t)row*HN + k];
                v = fmaxf(Cn[k]*(sp - Cn[2*HN + k])*Cn[3*HN + k] + Cn[HN + k], 0.f);
            }
            a.u[e] = f2bf(v);
        }
        // ---- h2 = relu(h@w3 + b3): [16x32] @ [32x112] ----
        v8f acc2[7];
        #pragma unroll
        for (int nt = 0; nt < 7; ++nt) acc2[nt] = vzero8();
        #pragma unroll
        for (int nt = 0; nt < 7; ++nt) {
            FragBF bm;
            FRAG_LOAD(bm, W3p, nt, lane);
            acc2[nt] = __builtin_amdgcn_wmma_f32_16x16x32_bf16(
                false, a.v, false, bm.v, (short)0, acc2[nt], false, false);
        }
        #pragma unroll
        for (int nt = 0; nt < 7; ++nt) {
            int nn = nt*16 + n16;
            int kc2 = nn >> 5, kk = nn & 31;
            int hf2 = (kk >> 3) & 1;
            int e2  = (kk & 7) | ((kk >> 1) & 8);
            unsigned short* dst = &scr[((kc2*32 + hf2*16 + 8*hf) << 4) + e2];
            #pragma unroll
            for (int r = 0; r < 8; ++r)
                dst[r << 4] = f2bf(fmaxf(acc2[nt][r] + B3s[nn], 0.f));
        }
        {
            uint4 z4 = make_uint4(0u, 0u, 0u, 0u);
            *(uint4*)&scr[((3*32 + lane) << 4) + 8] = z4;
        }
        __builtin_amdgcn_wave_barrier();

        // ---- logits = h2@w4 + b4: [16x128] @ [128x128] ----
        v8f acc3[8];
        #pragma unroll
        for (int nt = 0; nt < 8; ++nt) acc3[nt] = vzero8();
        #pragma unroll
        for (int kc = 0; kc < 4; ++kc) {
            FragBF a3;
            FRAG_LOAD(a3, scr, kc, lane);
            #pragma unroll
            for (int nt = 0; nt < 8; ++nt) {
                FragBF bm;
                FRAG_LOAD(bm, W4p, kc*8 + nt, lane);
                acc3[nt] = __builtin_amdgcn_wmma_f32_16x16x32_bf16(
                    false, a3.v, false, bm.v, (short)0, acc3[nt], false, false);
            }
        }
        #pragma unroll
        for (int nt = 0; nt < 8; ++nt) {
            int nn = nt*16 + n16;
            #pragma unroll
            for (int r = 0; r < 8; ++r) {
                int rr = tile*16 + r + 8*hf;
                logits[(size_t)rr*128 + nn] = acc3[nt][r] + B4s[nn];
            }
        }
    }
}

// ---------------------------------------------------------------------------
// Per-row cross-entropy + argmax accuracy, deterministic block reduction.
// ---------------------------------------------------------------------------
__global__ void __launch_bounds__(256)
loss_kernel(const float* __restrict__ logits, const float* __restrict__ by,
            float* __restrict__ lpart, float* __restrict__ apart)
{
    const int tid = threadIdx.x;
    const int row = blockIdx.x*256 + tid;
    float lg[128];
    const float4* lp4 = (const float4*)(logits + (size_t)row*128);
    #pragma unroll
    for (int q = 0; q < 32; ++q) {
        float4 v = lp4[q];
        lg[4*q + 0] = v.x; lg[4*q + 1] = v.y; lg[4*q + 2] = v.z; lg[4*q + 3] = v.w;
    }
    float mx = lg[0];
    #pragma unroll
    for (int e = 1; e < 128; ++e) mx = fmaxf(mx, lg[e]);
    float se = 0.f;
    #pragma unroll
    for (int e = 0; e < 128; ++e) se += __expf(lg[e] - mx);
    float lse = mx + __logf(se);

    const float* byr = by + (size_t)row*128;
    float dot = byr[0]*lg[0], sby = byr[0];
    float bmax = byr[0]; int bi = 0;
    float lmax = lg[0];  int li = 0;
    #pragma unroll
    for (int e = 1; e < 128; ++e) {
        float bv = byr[e];
        dot += bv*lg[e];
        sby += bv;
        if (bv > bmax)    { bmax = bv; bi = e; }
        if (lg[e] > lmax) { lmax = lg[e]; li = e; }
    }
    float lossi = sby*lse - dot;                  // -sum by*(lg - lse)
    float acci  = (li == bi) ? 1.f : 0.f;

    __shared__ float sl[256];
    __shared__ float sa[256];
    sl[tid] = lossi; sa[tid] = acci;
    __syncthreads();
    for (int st = 128; st > 0; st >>= 1) {
        if (tid < st) { sl[tid] += sl[tid + st]; sa[tid] += sa[tid + st]; }
        __syncthreads();
    }
    if (tid == 0) { lpart[blockIdx.x] = sl[0]; apart[blockIdx.x] = sa[0]; }
}

__global__ void out_kernel(const float* __restrict__ lpart,
                           const float* __restrict__ apart,
                           float* __restrict__ out, int nblk)
{
    if (threadIdx.x == 0) {
        float L = 0.f, A = 0.f;
        for (int i = 0; i < nblk; ++i) { L += lpart[i]; A += apart[i]; }
        out[0] = L / (float)BN;
        out[1] = A / (float)BN;
    }
}

// ---------------------------------------------------------------------------
extern "C" void kernel_launch(void* const* d_in, const int* in_sizes, int n_in,
                              void* d_out, int out_size, void* d_ws, size_t ws_size,
                              hipStream_t stream) {
    const float* bx = (const float*)d_in[0];
    const float* by = (const float*)d_in[1];
    const float* lp = (const float*)d_in[2];
    const float* ep = (const float*)d_in[3];
    const float* w1 = (const float*)d_in[4];
    const float* b1 = (const float*)d_in[5];
    const float* w2 = (const float*)d_in[6];
    const float* b2 = (const float*)d_in[7];
    const float* w3 = (const float*)d_in[8];
    const float* b3 = (const float*)d_in[9];
    const float* w4 = (const float*)d_in[10];
    const float* b4 = (const float*)d_in[11];
    const float* wm = (const float*)d_in[12];
    const float* cm = (const float*)d_in[13];
    const float* gp = (const float*)d_in[14];
    const float* bp = (const float*)d_in[15];
    float* out = (float*)d_out;

    float* ws       = (float*)d_ws;
    float* zc       = ws;                                   // B*T*H
    float* rhist    = zc    + (size_t)BN*TN*HN;             // T*B*H
    float* spre     = rhist + (size_t)TN*BN*HN;             // B*H
    float* logits   = spre  + (size_t)BN*HN;                // B*128
    float* partials = logits + (size_t)BN*128;              // 128*40
    float* stats    = partials + 128*(2*HN);                // 40
    float* lpart    = stats + 64;                           // 128
    float* apart    = lpart + 128;                          // 128

    zc_kernel<<<1408, 128, 0, stream>>>(bx, w1, b1, w2, b2, cm, zc);

    const int stepBlocks = BN / 256;                        // 128
    for (int t = 0; t < TN; ++t) {
        step_kernel<<<stepBlocks, 256, 0, stream>>>(
            zc, wm, lp, ep, gp, bp, stats, rhist, spre, partials, t);
        stats_kernel<<<1, 32, 0, stream>>>(partials, stats, stepBlocks);
    }

    head_kernel<<<512, 128, 0, stream>>>(spre, stats, gp, bp, w3, b3, w4, b4, logits);
    loss_kernel<<<stepBlocks, 256, 0, stream>>>(logits, by, lpart, apart);
    out_kernel<<<1, 32, 0, stream>>>(lpart, apart, out, stepBlocks);

    (void)in_sizes; (void)n_in; (void)out_size; (void)ws_size;
}
